// GNNLSTM_33157147525322
// MI455X (gfx1250) — matrix operand order repaired and checked by reading, not compiled, not run on hardware
//
#include <hip/hip_runtime.h>
#include <hip/hip_bf16.h>
#include <math.h>

// ---------------- types ----------------
typedef __attribute__((ext_vector_type(16))) __bf16 v16bf;
typedef __attribute__((ext_vector_type(8)))  __bf16 v8bf;
typedef __attribute__((ext_vector_type(8)))  float  v8f;

#define NN   100
#define CC   32
#define HH   512
#define TT   64
#define BB   16
#define EE   2000
#define HOR  10
#define BN   (BB*NN)          // 1600
#define NC   (NN*CC)          // 3200
#define G4H  (4*HH)           // 2048
#define TB   (TT*BB)          // 1024

__device__ __forceinline__ __bf16 f2bf(float x) {
    union { float f; unsigned u; } v; v.f = x;
    unsigned r = v.u + 0x7FFFu + ((v.u >> 16) & 1u);   // RNE
    union { unsigned short s; __bf16 b; } o; o.s = (unsigned short)(r >> 16);
    return o.b;
}
__device__ __forceinline__ float sigf(float x) { return 1.0f / (1.0f + expf(-x)); }
__device__ __forceinline__ float gelu_exact(float x) {
    return 0.5f * x * (1.0f + erff(x * 0.70710678118654752f));
}

// ---------------- utility kernels ----------------
__global__ void fill_zero_f32(float* __restrict__ p, int n) {
    int i = blockIdx.x * blockDim.x + threadIdx.x;
    if (i < n) p[i] = 0.0f;
}
__global__ void copy_f32(float* __restrict__ d, const float* __restrict__ s, int n) {
    int i = blockIdx.x * blockDim.x + threadIdx.x;
    if (i < n) d[i] = s[i];
}
__global__ void cvt_f32_bf16(__bf16* __restrict__ d, const float* __restrict__ s, int n) {
    int i = blockIdx.x * blockDim.x + threadIdx.x;
    if (i < n) d[i] = f2bf(s[i]);
}

// ---------------- graph normalization (base graph, identical for every copy) ----------------
__global__ void norm_kernel(const int* __restrict__ ei, const float* __restrict__ ew,
                            float* __restrict__ nrm) {
    __shared__ float s_deg[NN];
    __shared__ float s_dinv[NN];
    int tid = threadIdx.x;
    if (tid < NN) s_deg[tid] = 0.0f;
    __syncthreads();
    for (int e = tid; e < EE; e += blockDim.x)
        atomicAdd(&s_deg[ei[EE + e]], ew[e]);
    __syncthreads();
    if (tid < NN) {
        float d = s_deg[tid];
        s_dinv[tid] = (d > 0.0f) ? (1.0f / sqrtf(fmaxf(d, 1e-12f))) : 0.0f;
    }
    __syncthreads();
    for (int e = tid; e < EE; e += blockDim.x)
        nrm[e] = s_dinv[ei[e]] * ew[e] * s_dinv[ei[EE + e]];
}

// ---------------- ARMA conv (one graph-copy per block; LDS scatter-accumulate) ----------------
// mode 0: write f32 gelu result        -> outF[block*3200 + i]
// mode 1: write bf16 (gelu + W_embed)  -> outB[block*3200 + i]
__global__ void arma_kernel(const int* __restrict__ ei, const float* __restrict__ nrm,
                            const float* __restrict__ x,
                            const float* __restrict__ w, const float* __restrict__ v,
                            const float* __restrict__ bias,
                            const float* __restrict__ W_embed,
                            float* __restrict__ outF, __bf16* __restrict__ outB, int mode) {
    __shared__ float s_agg[NC];
    __shared__ float s_scale[EE];
    __shared__ int   s_col[EE];
    __shared__ float s_w[CC], s_v[CC], s_b[CC];
    int tid = threadIdx.x;
    int base = blockIdx.x * NN;
    for (int i = tid; i < NC; i += blockDim.x) s_agg[i] = 0.0f;
    if (tid < CC) { s_w[tid] = w[tid]; s_v[tid] = v[tid]; s_b[tid] = bias[tid]; }
    for (int e = tid; e < EE; e += blockDim.x) {
        s_col[e] = ei[EE + e];
        s_scale[e] = nrm[e] * x[base + ei[e]];
    }
    __syncthreads();
    for (int idx = tid; idx < EE * CC; idx += blockDim.x) {
        int e = idx >> 5, ch = idx & 31;
        atomicAdd(&s_agg[s_col[e] * CC + ch], s_scale[e] * s_w[ch]);
    }
    __syncthreads();
    for (int i = tid; i < NC; i += blockDim.x) {
        int n = i >> 5, ch = i & 31;
        float xv = x[base + n];
        float g = gelu_exact(s_agg[i] + xv * s_v[ch] + s_b[ch]);
        size_t o = (size_t)blockIdx.x * NC + i;
        if (mode == 0) outF[o] = g;
        else           outB[o] = f2bf(g + W_embed[i]);
    }
}

// ---------------- build LSTM layer-0 input (gather + embed + bf16) ----------------
__global__ void build_xall(const float* __restrict__ X1, const float* __restrict__ W_embed,
                           __bf16* __restrict__ Xall) {
    int idx = blockIdx.x * blockDim.x + threadIdx.x;
    if (idx >= TB * NC) return;
    int r  = idx / NC, nc = idx % NC;
    int t  = r >> 4, b = r & 15;
    // xs[t,b,n,c] = X1[((b*N+n)*T + t), c] + W_embed[n,c]   (reference reshape semantics)
    int n  = nc >> 5, c = nc & 31;
    float val = X1[((size_t)((b * NN + n) * TT + t) << 5) + c] + W_embed[nc];
    Xall[idx] = f2bf(val);
}

// ---------------- WMMA GEMM:  C(M,N) = A(M,K) * Bw(N,K)^T  (bf16 in, f32 acc) ----------------
// One wave computes a 16x64 output strip (4 N-tiles) so each A fragment feeds 4 WMMAs.
// grid.x = N/64, grid.y = M/16, block = 32 (one wave32; EXEC all ones)
__global__ __launch_bounds__(32)
void wmma_gemm_nt4(const __bf16* __restrict__ A, int lda,
                   const __bf16* __restrict__ Bw, int ldb,
                   float* __restrict__ C, int ldc, int K) {
    int lane = threadIdx.x;
    int nb = blockIdx.x << 6;
    int mb = blockIdx.y << 4;
    int lo = lane & 15;
    int hi = lane >> 4;                       // half-wave select
    // A 16x32 bf16 layout: lane=M; lanes0-15 K:{0..7,16..23}, lanes16-31 K:{8..15,24..31}
    const __bf16* arow  = A  + (size_t)(mb + lo) * lda + (hi << 3);
    // B 32x16 bf16 layout: lane=N; lanes0-15 K:{0..7,16..23}, lanes16-31 K:{8..15,24..31}
    const __bf16* brow  = Bw + (size_t)(nb + lo) * ldb + (hi << 3);
    v8f acc[4] = {v8f{}, v8f{}, v8f{}, v8f{}};
    for (int k0 = 0; k0 < K; k0 += 32) {
        v8bf a0 = *(const v8bf*)(arow + k0);
        v8bf a1 = *(const v8bf*)(arow + k0 + 16);
        if (k0 + 128 < K) __builtin_prefetch(arow + k0 + 128, 0, 1);
        v16bf a;
#pragma unroll
        for (int i = 0; i < 8; ++i) { a[i] = a0[i]; a[i + 8] = a1[i]; }
#pragma unroll
        for (int nt = 0; nt < 4; ++nt) {
            const __bf16* br = brow + ((size_t)nt << 4) * ldb;
            v8bf b0 = *(const v8bf*)(br + k0);
            v8bf b1 = *(const v8bf*)(br + k0 + 16);
            v16bf b;
#pragma unroll
            for (int i = 0; i < 8; ++i) { b[i] = b0[i]; b[i + 8] = b1[i]; }
            acc[nt] = __builtin_amdgcn_wmma_f32_16x16x32_bf16(false, a, false, b,
                                                              (short)0, acc[nt], false, false);
        }
    }
    // C/D 16x16 f32: lane=N (lo), VGPR i = row (mb + hi*8 + i)
#pragma unroll
    for (int nt = 0; nt < 4; ++nt) {
        float* crow = C + (size_t)(mb + (hi << 3)) * ldc + nb + (nt << 4) + lo;
#pragma unroll
        for (int i = 0; i < 8; ++i) crow[(size_t)i * ldc] = acc[nt][i];
    }
}

// ---------------- LSTM cell (torch gate order i,f,g,o); c updated in place ----------------
__global__ void lstm_cell_kernel(const float* __restrict__ gx, const float* __restrict__ gh,
                                 const float* __restrict__ bih, const float* __restrict__ bhh,
                                 float* __restrict__ c_state, float* __restrict__ h_f32,
                                 __bf16* __restrict__ h_bf16) {
    int idx = blockIdx.x * blockDim.x + threadIdx.x;
    if (idx >= BB * HH) return;
    int b = idx >> 9, u = idx & 511;
    int g = b * G4H;
    float gi = gx[g + u]            + gh[g + u]            + bih[u]            + bhh[u];
    float gf = gx[g + HH + u]       + gh[g + HH + u]       + bih[HH + u]       + bhh[HH + u];
    float gg = gx[g + 2 * HH + u]   + gh[g + 2 * HH + u]   + bih[2 * HH + u]   + bhh[2 * HH + u];
    float go = gx[g + 3 * HH + u]   + gh[g + 3 * HH + u]   + bih[3 * HH + u]   + bhh[3 * HH + u];
    float c = sigf(gf) * c_state[idx] + sigf(gi) * tanhf(gg);
    float h = sigf(go) * tanhf(c);
    c_state[idx] = c;
    h_f32[idx]   = h;
    h_bf16[idx]  = f2bf(h);
}

// ---------------- decoder projection: pred = h2 @ Wp^T + bp ; also scatter to d_out ----------------
__global__ void pred_kernel(const float* __restrict__ h2, const float* __restrict__ Wp,
                            const float* __restrict__ bp, float* __restrict__ pred_buf,
                            float* __restrict__ out, int j) {
    int idx = blockIdx.x * blockDim.x + threadIdx.x;
    if (idx >= BB * NN) return;
    int b = idx / NN, n = idx % NN;
    const float* hr = h2 + b * HH;
    const float* wr = Wp + n * HH;
    float s = bp[n];
    for (int k = 0; k < HH; ++k) s += hr[k] * wr[k];
    pred_buf[b * NN + n] = s;
    out[(size_t)(b * NN + n) * HOR + j] = s;
}

// ---------------- host orchestration ----------------
static inline int ceil_div(int a, int b) { return (a + b - 1) / b; }

extern "C" void kernel_launch(void* const* d_in, const int* in_sizes, int n_in,
                              void* d_out, int out_size, void* d_ws, size_t ws_size,
                              hipStream_t stream) {
    (void)in_sizes; (void)n_in; (void)out_size; (void)ws_size;
    const float* window   = (const float*)d_in[0];
    const int*   ei       = (const int*)d_in[1];
    const float* eattr    = (const float*)d_in[2];
    const float* W_embed  = (const float*)d_in[3];
    const float* tg_w     = (const float*)d_in[4];
    const float* tg_v     = (const float*)d_in[5];
    const float* tg_b     = (const float*)d_in[6];
    const float* gnn_w    = (const float*)d_in[7];
    const float* gnn_v    = (const float*)d_in[8];
    const float* gnn_b    = (const float*)d_in[9];
    const float* Wih0     = (const float*)d_in[10];
    const float* Whh0     = (const float*)d_in[11];
    const float* bih0     = (const float*)d_in[12];
    const float* bhh0     = (const float*)d_in[13];
    const float* Wih1     = (const float*)d_in[14];
    const float* Whh1     = (const float*)d_in[15];
    const float* bih1     = (const float*)d_in[16];
    const float* bhh1     = (const float*)d_in[17];
    const float* c1Wih    = (const float*)d_in[18];
    const float* c1Whh    = (const float*)d_in[19];
    const float* c1bih    = (const float*)d_in[20];
    const float* c1bhh    = (const float*)d_in[21];
    const float* c2Wih    = (const float*)d_in[22];
    const float* c2Whh    = (const float*)d_in[23];
    const float* c2bih    = (const float*)d_in[24];
    const float* c2bhh    = (const float*)d_in[25];
    const float* Wp       = (const float*)d_in[26];
    const float* bp       = (const float*)d_in[27];
    float* out = (float*)d_out;

    // ---- workspace carve-out ----
    char* wsp = (char*)d_ws;
    size_t off = 0;
    auto alloc = [&](size_t bytes) -> void* {
        void* p = wsp + off;
        off = (off + bytes + 255) & ~(size_t)255;
        return p;
    };
    float*  nrm      = (float*)alloc(EE * 4);
    float*  X1       = (float*)alloc((size_t)TB * NC * 4);         // 13.1 MB
    __bf16* Xall     = (__bf16*)alloc((size_t)TB * NC * 2);        //  6.6 MB
    __bf16* Wih0b    = (__bf16*)alloc((size_t)G4H * NC * 2);       // 13.1 MB
    __bf16* Whh0b    = (__bf16*)alloc((size_t)G4H * HH * 2);
    __bf16* Wih1b    = (__bf16*)alloc((size_t)G4H * HH * 2);
    __bf16* Whh1b    = (__bf16*)alloc((size_t)G4H * HH * 2);
    __bf16* c1Wihb   = (__bf16*)alloc((size_t)G4H * NC * 2);       // 13.1 MB
    __bf16* c1Whhb   = (__bf16*)alloc((size_t)G4H * HH * 2);
    __bf16* c2Wihb   = (__bf16*)alloc((size_t)G4H * HH * 2);
    __bf16* c2Whhb   = (__bf16*)alloc((size_t)G4H * HH * 2);
    float*  gx0      = (float*)alloc((size_t)TB * G4H * 4);        //  8.4 MB
    float*  gx1      = (float*)alloc((size_t)TB * G4H * 4);        //  8.4 MB
    __bf16* hs0      = (__bf16*)alloc((size_t)TB * HH * 2);
    __bf16* hs1      = (__bf16*)alloc((size_t)TB * HH * 2);
    float*  gh_buf   = (float*)alloc((size_t)BB * G4H * 4);
    __bf16* zero_h   = (__bf16*)alloc((size_t)BB * HH * 2);
    float*  cA       = (float*)alloc((size_t)BB * HH * 4);
    float*  cB       = (float*)alloc((size_t)BB * HH * 4);
    float*  hA       = (float*)alloc((size_t)BB * HH * 4);         // final h of layer0
    float*  hB       = (float*)alloc((size_t)BB * HH * 4);         // final h of layer1
    float*  d_c1     = (float*)alloc((size_t)BB * HH * 4);
    float*  d_c2     = (float*)alloc((size_t)BB * HH * 4);
    float*  d_h2     = (float*)alloc((size_t)BB * HH * 4);
    float*  d_h1f    = (float*)alloc((size_t)BB * HH * 4);
    __bf16* d_h1b    = (__bf16*)alloc((size_t)BB * HH * 2);
    __bf16* d_h2b    = (__bf16*)alloc((size_t)BB * HH * 2);
    float*  pred_buf = (float*)alloc((size_t)BN * 4);
    __bf16* xin_buf  = (__bf16*)alloc((size_t)BB * NC * 2);
    float*  gxd      = (float*)alloc((size_t)BB * G4H * 4);
    float*  ghd      = (float*)alloc((size_t)BB * G4H * 4);
    float*  gxd2     = (float*)alloc((size_t)BB * G4H * 4);
    float*  ghd2     = (float*)alloc((size_t)BB * G4H * 4);

    const int TPB = 256;
    // ---- init zeros (every call: deterministic) ----
    fill_zero_f32<<<ceil_div(BB * HH / 2, TPB), TPB, 0, stream>>>((float*)zero_h, BB * HH / 2);
    fill_zero_f32<<<ceil_div(BB * HH, TPB), TPB, 0, stream>>>(cA, BB * HH);
    fill_zero_f32<<<ceil_div(BB * HH, TPB), TPB, 0, stream>>>(cB, BB * HH);

    // ---- graph normalization (base graph) ----
    norm_kernel<<<1, 256, 0, stream>>>(ei, eattr, nrm);

    // ---- weight conversions to bf16 ----
    cvt_f32_bf16<<<ceil_div(G4H * NC, TPB), TPB, 0, stream>>>(Wih0b, Wih0, G4H * NC);
    cvt_f32_bf16<<<ceil_div(G4H * HH, TPB), TPB, 0, stream>>>(Whh0b, Whh0, G4H * HH);
    cvt_f32_bf16<<<ceil_div(G4H * HH, TPB), TPB, 0, stream>>>(Wih1b, Wih1, G4H * HH);
    cvt_f32_bf16<<<ceil_div(G4H * HH, TPB), TPB, 0, stream>>>(Whh1b, Whh1, G4H * HH);
    cvt_f32_bf16<<<ceil_div(G4H * NC, TPB), TPB, 0, stream>>>(c1Wihb, c1Wih, G4H * NC);
    cvt_f32_bf16<<<ceil_div(G4H * HH, TPB), TPB, 0, stream>>>(c1Whhb, c1Whh, G4H * HH);
    cvt_f32_bf16<<<ceil_div(G4H * HH, TPB), TPB, 0, stream>>>(c2Wihb, c2Wih, G4H * HH);
    cvt_f32_bf16<<<ceil_div(G4H * HH, TPB), TPB, 0, stream>>>(c2Whhb, c2Whh, G4H * HH);

    // ---- TGConv ARMA over T*B graph copies ----
    arma_kernel<<<TB, 256, 0, stream>>>(ei, nrm, window, tg_w, tg_v, tg_b,
                                        W_embed, X1, ((__bf16*)0), 0);
    // ---- build LSTM layer-0 input ----
    build_xall<<<ceil_div(TB * NC, TPB), TPB, 0, stream>>>(X1, W_embed, Xall);

    // ---- layer 0: hoisted input GEMM (1024x2048x3200), then recurrence ----
    {
        dim3 g(G4H / 64, TB / 16);
        wmma_gemm_nt4<<<g, 32, 0, stream>>>(Xall, NC, Wih0b, NC, gx0, G4H, NC);
    }
    for (int t = 0; t < TT; ++t) {
        const __bf16* hprev = (t == 0) ? zero_h : (hs0 + (size_t)(t - 1) * BB * HH);
        dim3 g(G4H / 64, 1);
        wmma_gemm_nt4<<<g, 32, 0, stream>>>(hprev, HH, Whh0b, HH, gh_buf, G4H, HH);
        lstm_cell_kernel<<<ceil_div(BB * HH, TPB), TPB, 0, stream>>>(
            gx0 + (size_t)t * BB * G4H, gh_buf, bih0, bhh0,
            cA, hA, hs0 + (size_t)t * BB * HH);
    }
    // ---- layer 1 ----
    {
        dim3 g(G4H / 64, TB / 16);
        wmma_gemm_nt4<<<g, 32, 0, stream>>>(hs0, HH, Wih1b, HH, gx1, G4H, HH);
    }
    for (int t = 0; t < TT; ++t) {
        const __bf16* hprev = (t == 0) ? zero_h : (hs1 + (size_t)(t - 1) * BB * HH);
        dim3 g(G4H / 64, 1);
        wmma_gemm_nt4<<<g, 32, 0, stream>>>(hprev, HH, Whh1b, HH, gh_buf, G4H, HH);
        lstm_cell_kernel<<<ceil_div(BB * HH, TPB), TPB, 0, stream>>>(
            gx1 + (size_t)t * BB * G4H, gh_buf, bih1, bhh1,
            cB, hB, hs1 + (size_t)t * BB * HH);
    }

    // ---- decoder init: carry = (h1f, h1f, h2f, h2f) ----
    copy_f32<<<ceil_div(BB * HH, TPB), TPB, 0, stream>>>(d_c1, hA, BB * HH);
    copy_f32<<<ceil_div(BB * HH, TPB), TPB, 0, stream>>>(d_c2, hB, BB * HH);
    copy_f32<<<ceil_div(BB * HH, TPB), TPB, 0, stream>>>(d_h2, hB, BB * HH);
    cvt_f32_bf16<<<ceil_div(BB * HH, TPB), TPB, 0, stream>>>(d_h1b, hA, BB * HH);
    cvt_f32_bf16<<<ceil_div(BB * HH, TPB), TPB, 0, stream>>>(d_h2b, hB, BB * HH);

    // ---- decoder: 10 autoregressive steps ----
    for (int j = 0; j < HOR; ++j) {
        pred_kernel<<<ceil_div(BN, TPB), TPB, 0, stream>>>(d_h2, Wp, bp, pred_buf, out, j);
        arma_kernel<<<BB, 256, 0, stream>>>(ei, nrm, pred_buf, gnn_w, gnn_v, gnn_b,
                                            W_embed, ((float*)0), xin_buf, 1);
        dim3 g(G4H / 64, 1);
        wmma_gemm_nt4<<<g, 32, 0, stream>>>(xin_buf, NC, c1Wihb, NC, gxd, G4H, NC);
        wmma_gemm_nt4<<<g, 32, 0, stream>>>(d_h1b, HH, c1Whhb, HH, ghd, G4H, HH);
        lstm_cell_kernel<<<ceil_div(BB * HH, TPB), TPB, 0, stream>>>(
            gxd, ghd, c1bih, c1bhh, d_c1, d_h1f, d_h1b);
        wmma_gemm_nt4<<<g, 32, 0, stream>>>(d_h1b, HH, c2Wihb, HH, gxd2, G4H, HH);
        wmma_gemm_nt4<<<g, 32, 0, stream>>>(d_h2b, HH, c2Whhb, HH, ghd2, G4H, HH);
        lstm_cell_kernel<<<ceil_div(BB * HH, TPB), TPB, 0, stream>>>(
            gxd2, ghd2, c2bih, c2bhh, d_c2, d_h2, d_h2b);
    }
}
